// softmax_readout_62251255988804
// MI455X (gfx1250) — matrix operand order repaired and checked by reading, not compiled, not run on hardware
//
#include <hip/hip_runtime.h>

// Problem sizes (fixed by reference)
#define N_ROWS 2048
#define DIM    2048
#define NCLS   50257
#define NCLSP  50304   // NCLS padded to multiple of 128 (K-pad for GEMM2, N-pad for GEMM1)

typedef __attribute__((ext_vector_type(16))) __bf16 v16bf;
typedef __attribute__((ext_vector_type(8)))  float  v8f;
typedef __attribute__((ext_vector_type(4)))  unsigned int u32x4;

union Frag16 {              // 16 bf16 = 32 bytes = one WMMA A/B operand per lane
    u32x4 q[2];
    v16bf v;
};

// ---------------- kernel 0a: x fp32 -> bf16 ----------------
__global__ __launch_bounds__(256)
void cvt_x_kernel(const float* __restrict__ x, __bf16* __restrict__ xbf) {
    int i = blockIdx.x * 256 + threadIdx.x;   // grid sized exactly
    xbf[i] = (__bf16)x[i];
}

// ---------------- kernel 0b: one read of W -> both bf16 layouts ----------------
// 64x64 fp32 tile async-copied to LDS, then emitted as:
//   wbf [DIM][NCLSP] row-major bf16 (zero col pad)  -> GEMM2 B operand
//   wt  [NCLSP][DIM] transposed bf16 (zero row pad) -> GEMM1 B operand
__global__ __launch_bounds__(256)
void transpose_w_kernel(const float* __restrict__ W, __bf16* __restrict__ wbf,
                        __bf16* __restrict__ wt) {
    __shared__ float ldsF[64 * 65];          // 65-dword row stride: conflict-free columns
    const int tid = threadIdx.x;
    const int k0  = blockIdx.x * 64;         // over DIM rows of W
    const int n0  = blockIdx.y * 64;         // over NCLSP cols of W
    const int nn  = tid & 63;
    const int kk4 = tid >> 6;                // 0..3

    const int  ncol = n0 + nn;
    const int  ncl  = (ncol < NCLS) ? ncol : (NCLS - 1);  // clamp addr (pad zeroed later)
    const bool inb  = (ncol < NCLS);

    // async copy tile into LDS (ASYNCcnt path); 4-byte ops dodge the odd row stride
    #pragma unroll
    for (int i = 0; i < 16; ++i) {
        int k = i * 4 + kk4;
        unsigned lds_off = (unsigned)(unsigned long long)&ldsF[k * 65 + nn];
        const float* gp  = W + (size_t)(k0 + k) * NCLS + ncl;
        asm volatile("global_load_async_to_lds_b32 %0, %1, off"
                     :: "v"(lds_off), "v"((unsigned long long)gp) : "memory");
    }
    asm volatile("s_wait_asynccnt 0x0" ::: "memory");
    __syncthreads();

    // row-major bf16 copy (coalesced along n)
    #pragma unroll
    for (int i = 0; i < 16; ++i) {
        int k   = i * 4 + kk4;
        float v = ldsF[k * 65 + nn];
        v = inb ? v : 0.f;
        wbf[(size_t)(k0 + k) * NCLSP + ncol] = (__bf16)v;
    }
    // transposed bf16 copy (coalesced along k; LDS column read is conflict-free)
    #pragma unroll
    for (int i = 0; i < 16; ++i) {
        int n   = i * 4 + kk4;
        float v = ldsF[nn * 65 + n];
        v = ((n0 + n) < NCLS) ? v : 0.f;
        wt[(size_t)(n0 + n) * DIM + k0 + nn] = (__bf16)v;
    }
}

// ---------------- kernel 1: logits = xbf @ W + b  (B from wt, no LDS) ----------------
// block = 128 threads = 4 waves; block tile 128(M) x 128(N); wave tile 64x64 (4x4 accums)
__global__ __launch_bounds__(128)
void gemm1_kernel(const __bf16* __restrict__ xbf, const __bf16* __restrict__ wt,
                  const float* __restrict__ bias, float* __restrict__ logits) {
    const int tid   = threadIdx.x;
    const int lane  = tid & 31;
    const int wave  = tid >> 5;
    const int lr    = lane & 15;      // column within 16
    const int lh    = lane >> 4;      // lane-half selects K grouping
    const int waveM = (wave >> 1) * 64;
    const int waveN = (wave & 1) * 64;
    const int blockM = blockIdx.x * 128;
    const int blockN = blockIdx.y * 128;   // tiles end exactly at NCLSP

    v8f acc[4][4] = {};

    for (int kb = 0; kb < DIM; kb += 32) {
        Frag16 a[4], b[4];
        // A fragments (16-bit A layout): lanes<16 hold K{0..7,16..23}, lanes>=16 +8
        #pragma unroll
        for (int ii = 0; ii < 4; ++ii) {
            const __bf16* ap = xbf + (size_t)(blockM + waveM + ii * 16 + lr) * DIM
                                   + kb + lh * 8;
            a[ii].q[0] = *(const u32x4*)ap;
            a[ii].q[1] = *(const u32x4*)(ap + 16);
        }
        // B fragments: lane = logits column n; wt row n is contiguous in k
        #pragma unroll
        for (int jj = 0; jj < 4; ++jj) {
            int n = blockN + waveN + jj * 16 + lr;
            const __bf16* bp = wt + (size_t)n * DIM + kb + lh * 8;
            b[jj].q[0] = *(const u32x4*)bp;
            b[jj].q[1] = *(const u32x4*)(bp + 16);
        }
        #pragma unroll
        for (int ii = 0; ii < 4; ++ii)
            #pragma unroll
            for (int jj = 0; jj < 4; ++jj)
                acc[ii][jj] = __builtin_amdgcn_wmma_f32_16x16x32_bf16(
                    false, a[ii].v, false, b[jj].v, (short)0, acc[ii][jj],
                    false, false);
    }

    // store logits + bias; C/D layout: VGPR v -> M = v + 8*lh, N = lr
    #pragma unroll
    for (int jj = 0; jj < 4; ++jj) {
        int n = blockN + waveN + jj * 16 + lr;
        if (n >= NCLS) continue;
        float bv = bias[n];
        #pragma unroll
        for (int ii = 0; ii < 4; ++ii) {
            #pragma unroll
            for (int v = 0; v < 8; ++v) {
                int m = blockM + waveM + ii * 16 + lh * 8 + v;
                logits[(size_t)m * NCLS + n] = acc[ii][jj][v] + bv;
            }
        }
    }
}

// ---------------- kernel 2: per-row online max / sum-exp / sum(y) ----------------
__global__ __launch_bounds__(256)
void rowstats_kernel(const float* __restrict__ logits, const float* __restrict__ y,
                     float* __restrict__ lse, float* __restrict__ ysum) {
    const int row = blockIdx.x;
    const int tid = threadIdx.x;
    const float* lrow = logits + (size_t)row * NCLS;
    const float* yrow = y      + (size_t)row * NCLS;

    float m = -3.0e38f, s = 0.f, ys = 0.f;
    for (int c = tid; c < NCLS; c += 256) {
        float v  = lrow[c];
        float nm = fmaxf(m, v);
        s = s * __expf(m - nm) + __expf(v - nm);
        m = nm;
        ys += yrow[c];
    }
    __shared__ float sm[256], ss[256], sy[256];
    sm[tid] = m; ss[tid] = s; sy[tid] = ys;
    __syncthreads();
    for (int off = 128; off > 0; off >>= 1) {
        if (tid < off) {
            float m2 = sm[tid + off], s2 = ss[tid + off];
            float nm = fmaxf(sm[tid], m2);
            ss[tid] = ss[tid] * __expf(sm[tid] - nm) + s2 * __expf(m2 - nm);
            sm[tid] = nm;
            sy[tid] += sy[tid + off];
        }
        __syncthreads();
    }
    if (tid == 0) {
        lse[row]  = sm[0] + __logf(ss[0]);   // max + log(sumexp)
        ysum[row] = sy[0];
    }
}

// ---------------- kernel 3: dlogits = ysum*softmax(logits) - y  (bf16, K-padded) ----------------
__global__ __launch_bounds__(256)
void dlogits_kernel(const float* __restrict__ logits, const float* __restrict__ y,
                    const float* __restrict__ lse, const float* __restrict__ ysum,
                    __bf16* __restrict__ dl) {
    const int row = blockIdx.y;
    const int c   = blockIdx.x * 256 + threadIdx.x;
    if (c >= NCLSP) return;
    float v = 0.f;                                   // zero-fill K padding
    if (c < NCLS) {
        float p = __expf(logits[(size_t)row * NCLS + c] - lse[row]);
        v = ysum[row] * p - y[(size_t)row * NCLS + c];
    }
    dl[(size_t)row * NCLSP + c] = (__bf16)v;
}

// ---------------- kernel 4: dx = dlogits @ W^T  (B from wbf rows, no LDS) ----------------
__global__ __launch_bounds__(128)
void gemm2_kernel(const __bf16* __restrict__ dl, const __bf16* __restrict__ wbf,
                  float* __restrict__ dx) {
    const int tid   = threadIdx.x;
    const int lane  = tid & 31;
    const int wave  = tid >> 5;
    const int lr    = lane & 15;
    const int lh    = lane >> 4;
    const int waveM = (wave >> 1) * 64;
    const int waveN = (wave & 1) * 64;
    const int blockM = blockIdx.x * 128;   // over N_ROWS
    const int blockN = blockIdx.y * 128;   // over DIM

    v8f acc[4][4] = {};

    for (int kb = 0; kb < NCLSP; kb += 32) {
        Frag16 a[4], b[4];
        #pragma unroll
        for (int ii = 0; ii < 4; ++ii) {
            const __bf16* ap = dl + (size_t)(blockM + waveM + ii * 16 + lr) * NCLSP
                                  + kb + lh * 8;
            a[ii].q[0] = *(const u32x4*)ap;
            a[ii].q[1] = *(const u32x4*)(ap + 16);
        }
        #pragma unroll
        for (int jj = 0; jj < 4; ++jj) {
            int j = blockN + waveN + jj * 16 + lr;
            const __bf16* bp = wbf + (size_t)j * NCLSP + kb + lh * 8;
            b[jj].q[0] = *(const u32x4*)bp;
            b[jj].q[1] = *(const u32x4*)(bp + 16);
        }
        #pragma unroll
        for (int ii = 0; ii < 4; ++ii)
            #pragma unroll
            for (int jj = 0; jj < 4; ++jj)
                acc[ii][jj] = __builtin_amdgcn_wmma_f32_16x16x32_bf16(
                    false, a[ii].v, false, b[jj].v, (short)0, acc[ii][jj],
                    false, false);
    }

    #pragma unroll
    for (int jj = 0; jj < 4; ++jj) {
        int j = blockN + waveN + jj * 16 + lr;
        #pragma unroll
        for (int ii = 0; ii < 4; ++ii) {
            #pragma unroll
            for (int v = 0; v < 8; ++v) {
                int m = blockM + waveM + ii * 16 + lh * 8 + v;
                dx[(size_t)m * DIM + j] = acc[ii][jj][v];
            }
        }
    }
}

extern "C" void kernel_launch(void* const* d_in, const int* in_sizes, int n_in,
                              void* d_out, int out_size, void* d_ws, size_t ws_size,
                              hipStream_t stream) {
    const float* x = (const float*)d_in[0];
    const float* y = (const float*)d_in[1];
    const float* W = (const float*)d_in[2];
    const float* b = (const float*)d_in[3];

    float* dx     = (float*)d_out;                      // [N_ROWS, DIM]
    float* logits = dx + (size_t)N_ROWS * DIM;          // [N_ROWS, NCLS]

    // workspace layout
    char* ws = (char*)d_ws;
    __bf16* dl  = (__bf16*)ws;                          // [N_ROWS][NCLSP] bf16
    size_t off  = (size_t)N_ROWS * NCLSP * sizeof(__bf16);
    __bf16* wbf = (__bf16*)(ws + off);                  // [DIM][NCLSP] bf16 (zero pad)
    off += (size_t)DIM * NCLSP * sizeof(__bf16);
    __bf16* wt  = (__bf16*)(ws + off);                  // [NCLSP][DIM] bf16 (zero pad)
    off += (size_t)NCLSP * DIM * sizeof(__bf16);
    __bf16* xbf = (__bf16*)(ws + off);                  // [N_ROWS][DIM] bf16
    off += (size_t)N_ROWS * DIM * sizeof(__bf16);
    float* lse  = (float*)(ws + off);                   // [N_ROWS]
    float* ysum = lse + N_ROWS;                         // [N_ROWS]

    cvt_x_kernel<<<(N_ROWS * DIM) / 256, 256, 0, stream>>>(x, xbf);

    transpose_w_kernel<<<dim3(DIM / 64, NCLSP / 64), 256, 0, stream>>>(W, wbf, wt);

    gemm1_kernel<<<dim3(N_ROWS / 128, NCLSP / 128), 128, 0, stream>>>(
        xbf, wt, b, logits);

    rowstats_kernel<<<N_ROWS, 256, 0, stream>>>(logits, y, lse, ysum);

    dlogits_kernel<<<dim3((NCLSP + 255) / 256, N_ROWS), 256, 0, stream>>>(
        logits, y, lse, ysum, dl);

    gemm2_kernel<<<dim3(N_ROWS / 128, DIM / 128), 128, 0, stream>>>(dl, wbf, dx);
}
